// CTC_kd_mtl_26886495273803
// MI455X (gfx1250) — matrix operand order repaired and checked by reading, not compiled, not run on hardware
//
#include <hip/hip_runtime.h>
#include <hip/hip_bf16.h>

// ---------------------------------------------------------------------------
// Problem constants (from reference setup_inputs)
// ---------------------------------------------------------------------------
#define B_  16
#define T_  512
#define E_  512
#define V_  5000
#define VP_ 5120            // V padded to multiple of 128 (block col-tile)
#define L_  64
#define S_  129             // 2L+1
#define ROWS_ (B_ * T_)     // 8192
#define NEG_ (-1e30f)

typedef __attribute__((ext_vector_type(16))) __bf16 v16bf;
typedef __attribute__((ext_vector_type(8)))  __bf16 v8bf;
typedef __attribute__((ext_vector_type(8)))  float  v8f;

// float -> bf16 (round to nearest even), stored as raw u16 bits
__device__ __forceinline__ unsigned short f2bf(float f) {
    unsigned int u = __float_as_uint(f);
    unsigned int r = u + 0x7FFFu + ((u >> 16) & 1u);
    return (unsigned short)(r >> 16);
}

// ---------------------------------------------------------------------------
// Kernel 0a: hs_pad [8192 x 512] f32 -> bf16 (same layout)
// ---------------------------------------------------------------------------
__global__ void conv_hs_kernel(const float* __restrict__ hs,
                               unsigned short* __restrict__ out, int n) {
    int gid = blockIdx.x * blockDim.x + threadIdx.x;
    if (gid < n) out[gid] = f2bf(hs[gid]);
}

// ---------------------------------------------------------------------------
// Kernel 0b: W [512 x 5000] -> Wt_bf [5120 x 512] bf16 (transposed, padded)
// ---------------------------------------------------------------------------
__global__ void conv_wt_kernel(const float* __restrict__ W,
                               unsigned short* __restrict__ out) {
    int gid = blockIdx.x * blockDim.x + threadIdx.x;   // over 5120*512
    if (gid >= VP_ * E_) return;
    int n = gid >> 9;          // column of W (0..5119)
    int k = gid & (E_ - 1);    // row of W
    float v = (n < V_) ? W[k * V_ + n] : 0.0f;
    out[gid] = f2bf(v);
}

// ---------------------------------------------------------------------------
// Kernel 1: blocked WMMA GEMM  logits[8192 x 5120] = hs_bf @ Wt_bf^T + bias
//   Block = 256 threads (8 waves) computing a 64x128 output tile:
//     wave (wr 0..3, wc 0..1) computes a 16x64 strip = 4 accumulators that
//     share one A fragment per K-step (4x A reuse in registers).
//   A tile (64 rows x 512 K bf16 = 64 KB) staged once per block in LDS and
//   consumed via ds_load_b128; B streamed from L2 via global_load_b128.
//   K=512 in steps of 32 via v_wmma_f32_16x16x32_bf16 (f32 accumulate).
//   Grid divides exactly (VP_ multiple of 128) -> EXEC all-1s everywhere.
//   Fragment layouts per CDNA5 ISA 7.12.2:
//     A (16x32 bf16): lane m=lane%16; K runs [hi*8..+7] and [16+hi*8..+7]
//     B (32x16 bf16): lane n=lane%16; K runs [hi*8..+7] and [16+hi*8..+7]
//     C (16x16 f32):  vgpr r -> M = r + 8*hi, N = lane%16
// ---------------------------------------------------------------------------
__global__ void __launch_bounds__(256)
gemm_wmma_kernel(const unsigned short* __restrict__ hsbf,
                 const unsigned short* __restrict__ wtbf,
                 const float* __restrict__ bias,
                 float* __restrict__ logits) {
    __shared__ unsigned short lds_a[64 * E_];   // 64 KB

    const int tid = threadIdx.x;
    const int lane = tid & 31;
    const int wave = tid >> 5;
    const int wr = wave >> 1;                   // row tile in block: 0..3
    const int wc = wave & 1;                    // col strip in block: 0..1
    const int row0 = blockIdx.x * 64;
    const int colbase = blockIdx.y * 128 + wc * 64;

    // cooperative stage of A tile: 64 consecutive rows = flat 64 KB copy
    {
        const uint4* __restrict__ src = (const uint4*)(hsbf + (size_t)row0 * E_);
        uint4* dst = (uint4*)lds_a;
#pragma unroll
        for (int i = 0; i < 16; ++i)
            dst[i * 256 + tid] = src[i * 256 + tid];
    }
    __syncthreads();

    const int hi = lane >> 4;                   // 0 or 1
    const int nl = lane & 15;
    const unsigned short* aptr  = lds_a + (wr * 16 + nl) * E_ + (hi << 3);
    const unsigned short* bptr0 = wtbf + (size_t)(colbase + nl) * E_ + (hi << 3);

    v8f acc[4] = {{}, {}, {}, {}};
#pragma unroll
    for (int k0 = 0; k0 < E_; k0 += 32) {
        v8bf alo = *(const v8bf*)(aptr + k0);        // K = k0+hi*8 .. +7
        v8bf ahi = *(const v8bf*)(aptr + k0 + 16);   // K = k0+16+hi*8 .. +7
        v16bf a;
#pragma unroll
        for (int i = 0; i < 8; ++i) { a[i] = alo[i]; a[i + 8] = ahi[i]; }

#pragma unroll
        for (int ct = 0; ct < 4; ++ct) {
            const unsigned short* bp = bptr0 + (size_t)ct * 16 * E_ + k0;
            v8bf blo = *(const v8bf*)(bp);           // K = k0+hi*8 .. +7
            v8bf bhi = *(const v8bf*)(bp + 16);      // K = k0+16+hi*8 .. +7
            v16bf bm;
#pragma unroll
            for (int i = 0; i < 8; ++i) { bm[i] = blo[i]; bm[i + 8] = bhi[i]; }
            acc[ct] = __builtin_amdgcn_wmma_f32_16x16x32_bf16(
                /*neg_a=*/false, a, /*neg_b=*/false, bm,
                /*c_mod=*/(short)0, acc[ct],
                /*reuse_a=*/false, /*reuse_b=*/false);
        }
    }

#pragma unroll
    for (int ct = 0; ct < 4; ++ct) {
        const int ncol = colbase + ct * 16 + nl;
        const float bv = (ncol < V_) ? bias[ncol] : 0.0f;
#pragma unroll
        for (int r = 0; r < 8; ++r) {
            int row = row0 + wr * 16 + r + (hi << 3);
            logits[(size_t)row * VP_ + ncol] = acc[ct][r] + bv;
        }
    }
}

// ---------------------------------------------------------------------------
// Block reduction helpers (blockDim.x == 256)
// ---------------------------------------------------------------------------
__device__ __forceinline__ float blockMax256(float v, float* sh) {
    sh[threadIdx.x] = v; __syncthreads();
    for (int o = 128; o > 0; o >>= 1) {
        if (threadIdx.x < o) sh[threadIdx.x] = fmaxf(sh[threadIdx.x], sh[threadIdx.x + o]);
        __syncthreads();
    }
    float r = sh[0]; __syncthreads();
    return r;
}
__device__ __forceinline__ float blockSum256(float v, float* sh) {
    sh[threadIdx.x] = v; __syncthreads();
    for (int o = 128; o > 0; o >>= 1) {
        if (threadIdx.x < o) sh[threadIdx.x] += sh[threadIdx.x + o];
        __syncthreads();
    }
    float r = sh[0]; __syncthreads();
    return r;
}

// ---------------------------------------------------------------------------
// Kernel 2: per-row log-softmax stats, KD cross-entropy, CTC emit gather.
//   One 256-thread block per (b,t) row (8192 blocks).
//   cxe[b,t] = logZ_logits - sum_j p_j * logit_j   (0 if t >= hlen[b])
//   emit[b,t,s] = logit[ext_s] - logZ_logits
// ---------------------------------------------------------------------------
__global__ void row_stats_kernel(const float* __restrict__ logits,
                                 const float* __restrict__ soft,
                                 const int* __restrict__ hlens,
                                 const int* __restrict__ targets,
                                 float* __restrict__ cxe,
                                 float* __restrict__ emit) {
    __shared__ float sred[256];
    __shared__ float sLogZ;
    const int row = blockIdx.x;           // 0..8191
    const int b = row >> 9;               // row / T_
    const int t = row & (T_ - 1);
    const int tid = threadIdx.x;
    const float* lrow = logits + (size_t)row * VP_;
    const float* srow = soft + (size_t)row * V_;

    // pass 1: maxes
    float ml = NEG_, ms = NEG_;
    for (int c = tid; c < V_; c += 256) {
        ml = fmaxf(ml, lrow[c]);
        ms = fmaxf(ms, srow[c]);
    }
    ml = blockMax256(ml, sred);
    ms = blockMax256(ms, sred);

    // pass 2: sums
    float zl = 0.0f, zs = 0.0f, sp = 0.0f;
    for (int c = tid; c < V_; c += 256) {
        float l = lrow[c];
        float p = expf(srow[c] - ms);
        zl += expf(l - ml);
        zs += p;
        sp += p * l;
    }
    zl = blockSum256(zl, sred);
    zs = blockSum256(zs, sred);
    sp = blockSum256(sp, sred);

    if (tid == 0) {
        float logZ = ml + logf(zl);
        sLogZ = logZ;
        cxe[row] = (t < hlens[b]) ? (logZ - sp / zs) : 0.0f;
    }
    __syncthreads();

    if (tid < S_) {
        int cls = (tid & 1) ? targets[b * L_ + (tid >> 1)] : 0;  // blank=0
        emit[(size_t)row * S_ + tid] = lrow[cls] - sLogZ;
    }
}

// ---------------------------------------------------------------------------
// Kernel 3: CTC alpha recursion. One block per batch element (16 blocks),
//   S=129 states in LDS, sequential over frames, one barrier per frame.
// ---------------------------------------------------------------------------
__global__ void ctc_alpha_kernel(const float* __restrict__ emit,
                                 const int* __restrict__ hlens,
                                 const int* __restrict__ targets,
                                 const int* __restrict__ olens,
                                 float* __restrict__ ctc_out) {
    __shared__ float A0[S_], A1[S_];
    __shared__ int ext[S_];
    __shared__ unsigned char allow[S_];
    const int b = blockIdx.x;
    const int tid = threadIdx.x;
    const float* em = emit + (size_t)b * T_ * S_;
    const int hlen = hlens[b];

    if (tid < S_) ext[tid] = (tid & 1) ? targets[b * L_ + (tid >> 1)] : 0;
    __syncthreads();
    if (tid < S_) {
        allow[tid] = (tid >= 2) && (ext[tid] != 0) && (ext[tid] != ext[tid - 2]);
        A0[tid] = (tid == 0) ? em[0] : (tid == 1) ? em[1] : NEG_;
    }
    __syncthreads();

    float* cur = A0;
    float* nxt = A1;
    for (int t = 1; t < hlen; ++t) {
        if (tid < S_) {
            float a  = cur[tid];
            float s1 = (tid >= 1) ? cur[tid - 1] : NEG_;
            float s2 = allow[tid] ? cur[tid - 2] : NEG_;
            float m = fmaxf(a, fmaxf(s1, s2));
            float e = em[(size_t)t * S_ + tid];
            nxt[tid] = m + logf(expf(a - m) + expf(s1 - m) + expf(s2 - m)) + e;
        }
        __syncthreads();
        float* tmp = cur; cur = nxt; nxt = tmp;
    }

    if (tid == 0) {
        int end = 2 * olens[b];
        float a1 = cur[end];
        float a2 = cur[end - 1];
        float m = fmaxf(a1, a2);
        float ll = m + logf(expf(a1 - m) + expf(a2 - m));
        float loss = -ll;
        if (loss > 1e20f) loss = 0.0f;   // zero_infinity
        ctc_out[b] = loss;
    }
}

// ---------------------------------------------------------------------------
// Kernel 4: final scalar: 0.5*kd_sum/B + 0.5*ctc_sum/B
// ---------------------------------------------------------------------------
__global__ void finalize_kernel(const float* __restrict__ cxe,
                                const float* __restrict__ ctcv,
                                float* __restrict__ out) {
    __shared__ float sred[256];
    const int tid = threadIdx.x;
    float s = 0.0f;
    for (int i = tid; i < ROWS_; i += 256) s += cxe[i];
    float kd = blockSum256(s, sred);
    float c = (tid < B_) ? ctcv[tid] : 0.0f;
    float ctc = blockSum256(c, sred);
    if (tid == 0)
        out[0] = 0.5f * kd / (float)B_ + 0.5f * ctc / (float)B_;
}

// ---------------------------------------------------------------------------
// Host launcher
// ---------------------------------------------------------------------------
extern "C" void kernel_launch(void* const* d_in, const int* in_sizes, int n_in,
                              void* d_out, int out_size, void* d_ws, size_t ws_size,
                              hipStream_t stream) {
    const float* hs_pad      = (const float*)d_in[0];  // [16,512,512]
    const float* soft_logits = (const float*)d_in[1];  // [16,512,5000]
    const float* W           = (const float*)d_in[2];  // [512,5000]
    const float* bias        = (const float*)d_in[3];  // [5000]
    const int*   hlens       = (const int*)d_in[4];    // [16]
    const int*   targets     = (const int*)d_in[5];    // [16,64]
    const int*   olens       = (const int*)d_in[6];    // [16]
    float* out = (float*)d_out;

    // workspace layout (bytes, 512-aligned)
    char* base = (char*)d_ws;
    size_t off = 0;
    auto alloc = [&](size_t bytes) { char* p = base + off; off = (off + bytes + 511) & ~((size_t)511); return p; };
    unsigned short* hs_bf = (unsigned short*)alloc((size_t)ROWS_ * E_ * 2);   //  8.0 MB
    unsigned short* wt_bf = (unsigned short*)alloc((size_t)VP_ * E_ * 2);     //  5.2 MB
    float* logits = (float*)alloc((size_t)ROWS_ * VP_ * 4);                   // 168 MB (~fits L2)
    float* cxe    = (float*)alloc((size_t)ROWS_ * 4);
    float* emit   = (float*)alloc((size_t)ROWS_ * S_ * 4);
    float* ctcv   = (float*)alloc((size_t)B_ * 4);

    // 0: fp32 -> bf16 conversions
    {
        int n = ROWS_ * E_;
        conv_hs_kernel<<<(n + 255) / 256, 256, 0, stream>>>(hs_pad, hs_bf, n);
        int m = VP_ * E_;
        conv_wt_kernel<<<(m + 255) / 256, 256, 0, stream>>>(W, wt_bf);
    }
    // 1: blocked WMMA GEMM (128 x 40 blocks of 64x128 output tile)
    {
        dim3 grid(ROWS_ / 64, VP_ / 128);
        gemm_wmma_kernel<<<grid, 256, 0, stream>>>(hs_bf, wt_bf, bias, logits);
    }
    // 2: per-row softmax stats + KD cxe + CTC emit gather
    row_stats_kernel<<<ROWS_, 256, 0, stream>>>(logits, soft_logits, hlens,
                                                targets, cxe, emit);
    // 3: CTC alpha recursion (one block per batch element)
    ctc_alpha_kernel<<<B_, 256, 0, stream>>>(emit, hlens, targets, olens, ctcv);
    // 4: final scalar
    finalize_kernel<<<1, 256, 0, stream>>>(cxe, ctcv, out);
}
